// SelfAttention_62577673502820
// MI455X (gfx1250) — matrix operand order repaired and checked by reading, not compiled, not run on hardware
//
#include <hip/hip_runtime.h>
#include <hip/hip_bf16.h>

// MI455X (gfx1250) flash self-attention, wave32 + v_wmma_f32_16x16x32_f16.
// B=8, C=256, E=64, N=4096. All GEMMs (projections, S^T=K@Q, P@V, output
// projection) run on the WMMA pipe; S is never materialized (flash softmax
// over the key index i, per reference softmax axis=1). K/V tiles are staged
// into per-wave LDS with double-buffered global_load_async_to_lds_b128
// (ASYNCcnt), and read back via explicit addrspace(3) ds_load_b128.

typedef __attribute__((ext_vector_type(16))) _Float16 v16h;
typedef __attribute__((ext_vector_type(8)))  _Float16 v8h;
typedef __attribute__((ext_vector_type(8)))  float    v8f;

// Explicit LDS (addrspace 3) vector types: 32-bit pointers, loads lower to
// ds_load_b128 (never FLAT). Offsets are raw LDS byte addresses; the dynamic
// shared segment starts at LDS offset 0 (no static __shared__ in the kernel),
// matching the byte offsets fed to the async-copy asm.
typedef __attribute__((address_space(3))) v8h  lds_v8h;
typedef __attribute__((address_space(3))) v16h lds_v16h;

#define BB 8
#define CC 256
#define EE 64
#define NN 4096

#define WMMA_F16(A, Bm, Cm) \
  __builtin_amdgcn_wmma_f32_16x16x32_f16(false, (A), false, (Bm), (short)0, (Cm), false, false)

// Load A/B-matrix lane data from global: two contiguous 8-half chunks (16B).
__device__ __forceinline__ v16h wload2(const _Float16* p0, const _Float16* p1) {
  v8h a = *(const v8h*)p0;
  v8h b = *(const v8h*)p1;
  v16h r;
#pragma unroll
  for (int i = 0; i < 8; ++i) { r[i] = a[i]; r[i + 8] = b[i]; }
  return r;
}

// Same, from LDS byte offsets (ds_load_b128 x2).
__device__ __forceinline__ v16h lds_load2x8(unsigned off0, unsigned off1) {
  v8h a = *(const lds_v8h*)off0;
  v8h b = *(const lds_v8h*)off1;
  v16h r;
#pragma unroll
  for (int i = 0; i < 8; ++i) { r[i] = a[i]; r[i + 8] = b[i]; }
  return r;
}

__device__ __forceinline__ v16h lds_load16(unsigned off) {  // 32B-aligned
  return *(const lds_v16h*)off;
}

// ---------------------------------------------------------------------------
// Kernel 0: one-shot weight conversion to f16 (equalized-lr scale folded in).
//   wh  = [3][E][C] f16  (w_key, w_query, w_value) * 1/sqrt(C)
//   wah = [C][E]    f16  w_att * 1/sqrt(E)
// ---------------------------------------------------------------------------
__global__ __launch_bounds__(256) void prep_weights(
    const float* __restrict__ wk, const float* __restrict__ wq,
    const float* __restrict__ wv, const float* __restrict__ wa,
    _Float16* __restrict__ wh, _Float16* __restrict__ wah) {
  const int idx = blockIdx.x * 256 + threadIdx.x;
  if (idx < 3 * EE * CC) {
    const float* src = (idx < EE * CC) ? wk : (idx < 2 * EE * CC ? wq : wv);
    const int r = idx & (EE * CC - 1);           // EE*CC = 16384, pow2
    wh[idx] = (_Float16)(src[r] * 0.0625f);
  }
  if (idx < CC * EE) {
    wah[idx] = (_Float16)(wa[idx] * 0.125f);
  }
}

// ---------------------------------------------------------------------------
// Kernel 1: fused projections. One wave per (batch, 16-column n-tile).
//   f,g -> [B][N][E] f16 (row-contiguous e, feeds S^T GEMM A/B loads)
//   hv  -> [B][E][N] f16 (row-contiguous n, feeds P*V B loads)
// ---------------------------------------------------------------------------
__global__ __launch_bounds__(128) void proj_kernel(
    const float* __restrict__ x,
    const _Float16* __restrict__ wh,
    const float* __restrict__ bk, const float* __restrict__ bq,
    const float* __restrict__ bv,
    _Float16* __restrict__ f, _Float16* __restrict__ g, _Float16* __restrict__ hv) {
  const int lane = threadIdx.x & 31;
  const int wid  = threadIdx.x >> 5;
  const int tile = blockIdx.x * 4 + wid;     // 0..2047
  const int b    = tile >> 8;
  const int n0   = (tile & 255) << 4;
  const int half = lane >> 4;                // 0: lanes 0-15, 1: lanes 16-31
  const int nl   = lane & 15;
  const int n    = n0 + nl;

  // x B-matrices: B[K=c(32), N=n(16)] per 32-wide c window; lane=col n,
  // rows c = cbase..cbase+15 (half selects K 0-15 vs 16-31).
  v16h Bx[8];
  const float* xb = x + (size_t)b * CC * NN;
#pragma unroll
  for (int kc = 0; kc < 8; ++kc) {
    const int cbase = kc * 32 + half * 16;
#pragma unroll
    for (int t = 0; t < 16; ++t)
      Bx[kc][t] = (_Float16)xb[(size_t)(cbase + t) * NN + n];
  }

  const float* Bs[3] = { bk, bq, bv };

#pragma unroll
  for (int p = 0; p < 3; ++p) {
    const _Float16* W = wh + (size_t)p * EE * CC;
    const float* Bias = Bs[p];
#pragma unroll
    for (int et = 0; et < 4; ++et) {
      const int erow = et * 16 + nl;         // A row (M=e) for this lane
      v8f acc = {};
#pragma unroll
      for (int kc = 0; kc < 8; ++kc) {
        // A[M=e, K=c]: halfs = c = k0..k0+15 contiguous (k0 = half*16)
        const _Float16* wr = W + (size_t)erow * CC + kc * 32 + half * 16;
        v16h A = wload2(wr, wr + 8);
        acc = WMMA_F16(A, Bx[kc], acc);
      }
      const int eoff = et * 16 + half * 8;   // e of VGPR row r=0
      if (p < 2) {                           // f/g: [N][E], 8 contiguous halfs
        _Float16* dst = (p == 0 ? f : g) + ((size_t)b * NN + n) * EE + eoff;
        v8h out;
#pragma unroll
        for (int r = 0; r < 8; ++r) out[r] = (_Float16)(acc[r] + Bias[eoff + r]);
        *(v8h*)dst = out;
      } else {                               // hv: [E][N]
#pragma unroll
        for (int r = 0; r < 8; ++r)
          hv[((size_t)b * EE + eoff + r) * NN + n] = (_Float16)(acc[r] + Bias[eoff + r]);
      }
    }
  }
}

// ---------------------------------------------------------------------------
// Kernel 2: flash attention. One wave per (batch, 16-query j-tile); loop over
// keys in steps of 32. S^T tiles (M=key i, N=query j) come out of WMMA in
// exactly the A-matrix lane layout needed for the P*V WMMA -> no transpose.
// K/V tiles staged per-wave in LDS via async-to-LDS, double buffered.
// ---------------------------------------------------------------------------

// Issue one 32-key tile (f: 32 rows x 128B, hv: 64 rows x 64B = 8KB total)
// as 16 x global_load_async_to_lds_b128 (each moves 16B per lane, 512B/wave).
__device__ __forceinline__ void issue_tile(
    const _Float16* fb, const _Float16* hvb, int i0,
    unsigned lds_byte_base, int lane) {
#pragma unroll
  for (int k = 0; k < 8; ++k) {                    // f part: [row 0..31][64 halfs]
    const int chunk = k * 32 + lane;               // 256 chunks of 16B
    const int row = chunk >> 3, cu = chunk & 7;
    const unsigned long long ga =
        (unsigned long long)(fb + (size_t)(i0 + row) * EE + cu * 8);
    const unsigned lo = lds_byte_base + (unsigned)chunk * 16u;
    asm volatile("global_load_async_to_lds_b128 %0, %1, off"
                 :: "v"(lo), "v"(ga) : "memory");
  }
#pragma unroll
  for (int k = 0; k < 8; ++k) {                    // hv part: [e 0..63][32 halfs]
    const int chunk = k * 32 + lane;
    const int e = chunk >> 2, c = chunk & 3;
    const unsigned long long ga =
        (unsigned long long)(hvb + (size_t)e * NN + i0 + c * 8);
    const unsigned lo = lds_byte_base + 4096u + (unsigned)chunk * 16u;
    asm volatile("global_load_async_to_lds_b128 %0, %1, off"
                 :: "v"(lo), "v"(ga) : "memory");
  }
}

__global__ __launch_bounds__(128) void attn_kernel(
    const _Float16* __restrict__ f, const _Float16* __restrict__ g,
    const _Float16* __restrict__ hv, _Float16* __restrict__ vatt) {
  // LDS: 64KB DYNAMIC shared (passed at launch). No static __shared__ here,
  // so the dynamic region starts at LDS byte offset 0 -- raw offsets below
  // are valid for both the async-copy writes and the AS(3) reads.
  // Layout: 4 waves x 2 buffers x 8KB; buffer = f tile [32][64h] + hv [64][32h].

  const int lane = threadIdx.x & 31;
  const int wid  = threadIdx.x >> 5;
  const int tile = blockIdx.x * 4 + wid;
  const int b    = tile >> 8;
  const int j0   = (tile & 255) << 4;
  const int half = lane >> 4;
  const int nl   = lane & 15;
  const int k0   = half * 8;

  // Q (g) as B[K=e, N=j], two 32-wide e windows, held in registers all loop.
  const _Float16* gq = g + ((size_t)b * NN + j0 + nl) * EE + half * 8;
  v16h Bg0 = wload2(gq, gq + 16);
  v16h Bg1 = wload2(gq + 32, gq + 48);

  const _Float16* fb  = f  + (size_t)b * NN * EE;
  const _Float16* hvb = hv + (size_t)b * EE * NN;

  const unsigned wbase = (unsigned)(wid * 16384);      // wave LDS region (bytes)

  v8f O[4] = {};                             // O[M=j, N=e], 4 e-tiles
  float m = -1e30f, l = 0.0f;

  issue_tile(fb, hvb, 0, wbase, lane);       // prologue: tile 0 in flight

  for (int it = 0; it < NN / 32; ++it) {
    const int i0 = it * 32;
    const unsigned fbase = wbase + (unsigned)((it & 1) * 8192);
    const unsigned vbase = fbase + 4096u;

    if (it + 1 < NN / 32) {
      issue_tile(fb, hvb, i0 + 32, wbase + (unsigned)(((it + 1) & 1) * 8192), lane);
      // async loads complete in order: <=16 outstanding => tile 'it' is in LDS
      asm volatile("s_wait_asynccnt 0x10" ::: "memory");
    } else {
      asm volatile("s_wait_asynccnt 0x0" ::: "memory");
    }

    // K (f) as A[M=i, K=e]; lo = keys i0..i0+15, hi = +16..31 (ds_load_b128)
    const unsigned rlo = fbase + (unsigned)(nl * 128 + k0 * 2);
    const unsigned rhi = rlo + 16u * 128u;
    v16h Alo0 = lds_load2x8(rlo,      rlo + 32);
    v16h Alo1 = lds_load2x8(rlo + 64, rlo + 96);
    v16h Ahi0 = lds_load2x8(rhi,      rhi + 32);
    v16h Ahi1 = lds_load2x8(rhi + 64, rhi + 96);

    v8f Slo = {}, Shi = {};
    Slo = WMMA_F16(Alo0, Bg0, Slo);
    Slo = WMMA_F16(Alo1, Bg1, Slo);
    Shi = WMMA_F16(Ahi0, Bg0, Shi);
    Shi = WMMA_F16(Ahi1, Bg1, Shi);

    // Online softmax over keys i for query j = j0+nl (lane pair L, L+16
    // jointly holds all 32 scores of one query; combine via xor-16 shuffle).
    float pm = -1e30f;
#pragma unroll
    for (int r = 0; r < 8; ++r) pm = fmaxf(pm, fmaxf(Slo[r], Shi[r]));
    pm = fmaxf(pm, __shfl_xor(pm, 16, 32));
    const float mnew = fmaxf(m, pm);
    const float corr = __expf(m - mnew);

    float psum = 0.0f;
    v16h P;                                  // already in A-matrix lane layout
#pragma unroll
    for (int r = 0; r < 8; ++r) {
      const float e0 = __expf(Slo[r] - mnew);
      const float e1 = __expf(Shi[r] - mnew);
      psum += e0 + e1;
      P[r]     = (_Float16)e0;               // K = k0 + r
      P[r + 8] = (_Float16)e1;               // K = k0 + 16 + r
    }
    psum += __shfl_xor(psum, 16, 32);
    l = l * corr + psum;
    m = mnew;

    // Rescale O: row M=j of O maps to VGPR r (j_rel = half*8 + r); broadcast
    // corr[j] from lane j_rel via ds_bpermute.
    float sc[8];
#pragma unroll
    for (int r = 0; r < 8; ++r) sc[r] = __shfl(corr, (half << 3) + r, 32);
#pragma unroll
    for (int t = 0; t < 4; ++t)
#pragma unroll
      for (int r = 0; r < 8; ++r) O[t][r] *= sc[r];

    // P @ V: B[K=i(32), N=e(16)] from LDS hv tile, two 16B chunks per lane.
#pragma unroll
    for (int t = 0; t < 4; ++t) {
      const unsigned vo = vbase + (unsigned)((t * 16 + nl) * 64 + half * 16);
      v16h Bv = lds_load2x8(vo, vo + 32);
      O[t] = WMMA_F16(P, Bv, O[t]);
    }
  }

  // Finalize: O /= l (broadcast per row), store as [E][N] f16.
  const float linv = 1.0f / l;
  float li[8];
#pragma unroll
  for (int r = 0; r < 8; ++r) li[r] = __shfl(linv, (half << 3) + r, 32);
#pragma unroll
  for (int t = 0; t < 4; ++t) {
    v8h out;
#pragma unroll
    for (int r = 0; r < 8; ++r) out[r] = (_Float16)(O[t][r] * li[r]);
    _Float16* dst = vatt + ((size_t)b * EE + t * 16 + nl) * NN + j0 + half * 8;
    *(v8h*)dst = out;                        // 8 contiguous halfs (16B)
  }
}

// ---------------------------------------------------------------------------
// Kernel 3: output projection (C x E) @ vatt + residual:
//   y = gamma[c] * (W_att/sqrt(E) @ v + b_att) + x.   2 wmma per c-tile.
// ---------------------------------------------------------------------------
__global__ __launch_bounds__(128) void out_kernel(
    const _Float16* __restrict__ vatt,
    const _Float16* __restrict__ wah, const float* __restrict__ ba,
    const float* __restrict__ gamma, const float* __restrict__ x,
    float* __restrict__ y) {
  const int lane = threadIdx.x & 31;
  const int wid  = threadIdx.x >> 5;
  const int tile = blockIdx.x * 4 + wid;
  const int b    = tile >> 8;
  const int n0   = (tile & 255) << 4;
  const int half = lane >> 4;
  const int nl   = lane & 15;
  const int n    = n0 + nl;
  const int k0   = half * 8;

  // vatt B-matrices B[K=e, N=n], two 32-wide e windows, loaded once.
  v16h Bv0, Bv1;
  const _Float16* vb = vatt + (size_t)b * EE * NN;
#pragma unroll
  for (int t = 0; t < 16; ++t) {
    Bv0[t] = vb[(size_t)(half * 16 + t) * NN + n];
    Bv1[t] = vb[(size_t)(32 + half * 16 + t) * NN + n];
  }

  for (int ct = 0; ct < 16; ++ct) {
    const int crow = ct * 16 + nl;           // A row (M=c)
    const _Float16* wr = wah + (size_t)crow * EE + half * 16;
    v16h A0 = wload2(wr,      wr + 8);
    v16h A1 = wload2(wr + 32, wr + 40);
    v8f acc = {};
    acc = WMMA_F16(A0, Bv0, acc);
    acc = WMMA_F16(A1, Bv1, acc);
#pragma unroll
    for (int r = 0; r < 8; ++r) {
      const int c = ct * 16 + half * 8 + r;
      const size_t idx = ((size_t)b * CC + c) * NN + n;
      y[idx] = gamma[c] * (acc[r] + ba[c]) + x[idx];
    }
  }
}

// ---------------------------------------------------------------------------
extern "C" void kernel_launch(void* const* d_in, const int* in_sizes, int n_in,
                              void* d_out, int out_size, void* d_ws, size_t ws_size,
                              hipStream_t stream) {
  const float* x  = (const float*)d_in[0];
  const float* wk = (const float*)d_in[1];
  const float* bk = (const float*)d_in[2];
  const float* wq = (const float*)d_in[3];
  const float* bq = (const float*)d_in[4];
  const float* wv = (const float*)d_in[5];
  const float* bv = (const float*)d_in[6];
  const float* wa = (const float*)d_in[7];
  const float* ba = (const float*)d_in[8];
  const float* gm = (const float*)d_in[9];
  float* y = (float*)d_out;

  // Workspace: f, g (N-major), hv, vatt (E-major): 4 x 4MB f16 = 16MB,
  // then f16 weights wh [3][E][C] (96KB) and wah [C][E] (32KB).
  _Float16* f   = (_Float16*)d_ws;
  _Float16* g   = f   + (size_t)BB * NN * EE;
  _Float16* hv  = g   + (size_t)BB * NN * EE;
  _Float16* va  = hv  + (size_t)BB * NN * EE;
  _Float16* wh  = va  + (size_t)BB * NN * EE;
  _Float16* wah = wh  + (size_t)3 * EE * CC;

  dim3 blk(128);                             // 4 waves / block (wave32)
  dim3 grid(BB * (NN / 16) / 4);             // 512 blocks, 2048 waves
  prep_weights<<<dim3((3 * EE * CC + 255) / 256), dim3(256), 0, stream>>>(
      wk, wq, wv, wa, wh, wah);
  proj_kernel<<<grid, blk, 0, stream>>>(x, wh, bk, bq, bv, f, g, hv);
  // 64KB dynamic LDS: 4 waves x 2 x 8KB double-buffered K/V tiles.
  attn_kernel<<<grid, blk, 65536, stream>>>(f, g, hv, va);
  out_kernel<<<grid, blk, 0, stream>>>(va, wah, ba, gm, x, y);
}
